// APPNP_58858231824467
// MI455X (gfx1250) — matrix operand order repaired and checked by reading, not compiled, not run on hardware
//
#include <hip/hip_runtime.h>
#include <stdint.h>

// Problem constants (from reference): IN=256, H=64, OUT=40, K_STEPS=10, ALPHA=0.1
#define IN_F   256
#define HID    64
#define OUT_F  40
#define OUT_P  48     // OUT padded to 3 WMMA col-tiles
#define KSTEPS 10
#define ALPHA  0.1f

typedef __attribute__((ext_vector_type(16))) __bf16 v16bf;
typedef __attribute__((ext_vector_type(8)))  float  v8f;

__device__ __forceinline__ v8f wmma_bf16(v16bf a, v16bf b, v8f c) {
  // (neg_a, A, neg_b, B, c_mod, C, reuse_a, reuse_b)
  return __builtin_amdgcn_wmma_f32_16x16x32_bf16(false, a, false, b, (short)0, c,
                                                 false, false);
}

__device__ __forceinline__ unsigned short f2bf(float f) {
  __bf16 h = (__bf16)f;
  return __builtin_bit_cast(unsigned short, h);
}

// A-matrix fragment (16x32 bf16) loaded from an f32 row. rowp -> start of the
// 32-wide K block. Lane group g holds K in [8g,8g+8) and [16+8g,16+8g+8).
__device__ __forceinline__ v16bf load_a_f32(const float* rowp, int g) {
  v16bf a;
  const float4* q0 = reinterpret_cast<const float4*>(rowp + 8 * g);
  const float4* q1 = reinterpret_cast<const float4*>(rowp + 16 + 8 * g);
  float4 f;
  f = q0[0]; a[0]=(__bf16)f.x; a[1]=(__bf16)f.y; a[2]=(__bf16)f.z; a[3]=(__bf16)f.w;
  f = q0[1]; a[4]=(__bf16)f.x; a[5]=(__bf16)f.y; a[6]=(__bf16)f.z; a[7]=(__bf16)f.w;
  f = q1[0]; a[8]=(__bf16)f.x; a[9]=(__bf16)f.y; a[10]=(__bf16)f.z; a[11]=(__bf16)f.w;
  f = q1[1]; a[12]=(__bf16)f.x; a[13]=(__bf16)f.y; a[14]=(__bf16)f.z; a[15]=(__bf16)f.w;
  return a;
}

// A-matrix fragment from bf16 storage (LDS or global), same blocked K pattern.
__device__ __forceinline__ v16bf load_a_bf16(const unsigned short* rowp, int g) {
  union { v16bf v; uint4 q[2]; } u;
  u.q[0] = *reinterpret_cast<const uint4*>(rowp + 8 * g);
  u.q[1] = *reinterpret_cast<const uint4*>(rowp + 16 + 8 * g);
  return u.v;
}

// B-matrix fragment: lane group g -> K in [8g,8g+8) and [16+8g,16+8g+8).
__device__ __forceinline__ v16bf load_b_bf16(const unsigned short* p, int g) {
  union { v16bf v; uint4 q[2]; } u;
  u.q[0] = *reinterpret_cast<const uint4*>(p + 8 * g);
  u.q[1] = *reinterpret_cast<const uint4*>(p + 16 + 8 * g);
  return u.v;
}

// ---------------- weight prep: transpose to [N][K] + convert to bf16 ---------
__global__ void prep_weights_kernel(const float* __restrict__ W1,
                                    const float* __restrict__ W2,
                                    const float* __restrict__ Wf,
                                    unsigned short* __restrict__ W1t,
                                    unsigned short* __restrict__ W2t,
                                    unsigned short* __restrict__ Wft) {
  int i = blockIdx.x * blockDim.x + threadIdx.x;
  if (i < HID * IN_F) {            // W1t[n][k] = W1[k][n]
    int n = i / IN_F, k = i % IN_F;
    W1t[i] = f2bf(W1[k * HID + n]);
  }
  if (i < HID * HID) {             // W2t[n][k] = W2[k][n]
    int n = i / HID, k = i % HID;
    W2t[i] = f2bf(W2[k * HID + n]);
  }
  if (i < OUT_P * HID) {           // Wft[n][k] = Wf[k][n], zero-padded n>=40
    int n = i / HID, k = i % HID;
    Wft[i] = (n < OUT_F) ? f2bf(Wf[k * OUT_F + n]) : (unsigned short)0;
  }
}

// ---------------- degree / dinv ----------------------------------------------
__global__ void deg_init_kernel(float* __restrict__ deg, int N) {
  int i = blockIdx.x * blockDim.x + threadIdx.x;
  if (i < N) deg[i] = 1.0f;        // self loop
}
__global__ void deg_acc_kernel(const long long* __restrict__ col,
                               float* __restrict__ deg, int E) {
  int e = blockIdx.x * blockDim.x + threadIdx.x;
  if (e < E) atomicAdd(&deg[(int)col[e]], 1.0f);
}
__global__ void dinv_kernel(float* __restrict__ deg, int N) {
  int i = blockIdx.x * blockDim.x + threadIdx.x;
  if (i < N) deg[i] = rsqrtf(deg[i]);   // deg >= 1 always (self loop)
}

// ---------------- fused MLP encode: h0 = relu(x@W1+b1)@W2+b2 -----------------
// 128 threads = 4 waves; block owns 16 rows; wave w owns cols [16w,16w+16).
// Requires N % 16 == 0 (true: 50000) and full EXEC for WMMA.
__global__ __launch_bounds__(128) void encode_kernel(
    const float* __restrict__ x,
    const unsigned short* __restrict__ W1t, const float* __restrict__ b1,
    const unsigned short* __restrict__ W2t, const float* __restrict__ b2,
    float* __restrict__ h0, int N) {
  __shared__ __align__(16) unsigned short hl[16 * HID];  // 16x64 bf16 tile
  const int lane = threadIdx.x & 31;
  const int wave = threadIdx.x >> 5;
  const int rowL = lane & 15;
  const int g    = lane >> 4;
  const int m0   = blockIdx.x * 16;
  const int nCol = wave * 16 + rowL;      // B col == C col for this lane
  const int aRow = m0 + rowL;

  // ---- GEMM1: x[16x256] @ W1[256x64] -> relu -> bf16 LDS tile
  v8f c = {};
  const float*          xrow = x   + (size_t)aRow * IN_F;
  const unsigned short* wrow = W1t + (size_t)nCol * IN_F;
#pragma unroll
  for (int k0 = 0; k0 < IN_F; k0 += 32) {
    v16bf a = load_a_f32(xrow + k0, g);
    v16bf b = load_b_bf16(wrow + k0, g);
    c = wmma_bf16(a, b, c);
  }
  {
    float bias = b1[nCol];
#pragma unroll
    for (int i = 0; i < 8; ++i) {
      float v = c[i] + bias;
      v = v > 0.f ? v : 0.f;
      hl[(g * 8 + i) * HID + nCol] = f2bf(v);   // row = i + 8*(lane>>4)
    }
  }
  __syncthreads();

  // ---- GEMM2: hl[16x64] @ W2[64x64] + b2 -> h0 (f32)
  v8f c2 = {};
  const unsigned short* w2row = W2t + (size_t)nCol * HID;
#pragma unroll
  for (int k0 = 0; k0 < HID; k0 += 32) {
    v16bf a = load_a_bf16(&hl[rowL * HID + k0], g);
    v16bf b = load_b_bf16(w2row + k0, g);
    c2 = wmma_bf16(a, b, c2);
  }
  {
    float bias = b2[nCol];
#pragma unroll
    for (int i = 0; i < 8; ++i)
      h0[(size_t)(m0 + g * 8 + i) * HID + nCol] = c2[i] + bias;
  }
}

// ---------------- propagation: scatter-add over edges ------------------------
// 256 threads = 4 edges x 64 feature lanes. h table (12.8MB) lives in L2.
__global__ __launch_bounds__(256) void edge_scatter_kernel(
    const long long* __restrict__ row, const long long* __restrict__ col,
    const float* __restrict__ dinv, const float* __restrict__ h,
    float* __restrict__ agg, int E) {
  int e = blockIdx.x * 4 + (threadIdx.x >> 6);
  int f = threadIdx.x & 63;
  if (e >= E) return;
  int r = (int)row[e];
  int cI = (int)col[e];
  float w = dinv[r] * dinv[cI];
  atomicAdd(&agg[(size_t)cI * HID + f], w * h[(size_t)r * HID + f]);
}

// hn = 0.9*(agg + dinv^2 * h) + 0.1*h0 ; re-zero agg for the next step
__global__ __launch_bounds__(256) void combine_kernel(
    const float* __restrict__ h, const float* __restrict__ h0,
    const float* __restrict__ dinv, float* __restrict__ agg,
    float* __restrict__ hn, int N) {
  int i = blockIdx.x * blockDim.x + threadIdx.x;
  if (i >= N * HID) return;
  int node = i >> 6;  // /HID
  float d = dinv[node];
  float a = agg[i];
  agg[i] = 0.f;
  hn[i] = (1.f - ALPHA) * (a + d * d * h[i]) + ALPHA * h0[i];
}

// ---------------- decode: out = h@Wf + bf ------------------------------------
// 96 threads = 3 waves covering 48 padded cols; store masked to 40.
__global__ __launch_bounds__(96) void decode_kernel(
    const float* __restrict__ h, const unsigned short* __restrict__ Wft,
    const float* __restrict__ bfv, float* __restrict__ out, int N) {
  const int lane = threadIdx.x & 31;
  const int wave = threadIdx.x >> 5;
  const int rowL = lane & 15;
  const int g    = lane >> 4;
  const int m0   = blockIdx.x * 16;
  const int nCol = wave * 16 + rowL;
  const int aRow = m0 + rowL;

  v8f c = {};
  const float*          hrow = h   + (size_t)aRow * HID;
  const unsigned short* wrow = Wft + (size_t)nCol * HID;
#pragma unroll
  for (int k0 = 0; k0 < HID; k0 += 32) {
    v16bf a = load_a_f32(hrow + k0, g);
    v16bf b = load_b_bf16(wrow + k0, g);
    c = wmma_bf16(a, b, c);
  }
  if (nCol < OUT_F) {
    float bias = bfv[nCol];
#pragma unroll
    for (int i = 0; i < 8; ++i)
      out[(size_t)(m0 + g * 8 + i) * OUT_F + nCol] = c[i] + bias;
  }
}

// =============================================================================
extern "C" void kernel_launch(void* const* d_in, const int* in_sizes, int n_in,
                              void* d_out, int out_size, void* d_ws, size_t ws_size,
                              hipStream_t stream) {
  (void)n_in; (void)out_size; (void)ws_size;
  const float*     x   = (const float*)d_in[0];
  const long long* ei  = (const long long*)d_in[1];   // int64 [2, E]
  const float*     W1  = (const float*)d_in[2];
  const float*     b1  = (const float*)d_in[3];
  const float*     W2  = (const float*)d_in[4];
  const float*     b2  = (const float*)d_in[5];
  const float*     Wf  = (const float*)d_in[6];
  const float*     bfv = (const float*)d_in[7];
  float* out = (float*)d_out;

  const int N = in_sizes[0] / IN_F;   // 50000 (divisible by 16)
  const int E = in_sizes[1] / 2;      // 800000
  const long long* rowIdx = ei;
  const long long* colIdx = ei + E;

  // workspace layout
  char* ws = (char*)d_ws;
  size_t off = 0;
  auto alloc = [&](size_t bytes) -> void* {
    void* p = ws + off;
    off = (off + bytes + 255) & ~(size_t)255;
    return p;
  };
  float* deg = (float*)alloc((size_t)N * 4);             // becomes dinv
  float* h0  = (float*)alloc((size_t)N * HID * 4);
  float* hA  = (float*)alloc((size_t)N * HID * 4);
  float* hB  = (float*)alloc((size_t)N * HID * 4);
  float* agg = (float*)alloc((size_t)N * HID * 4);
  unsigned short* W1t = (unsigned short*)alloc((size_t)HID * IN_F * 2);
  unsigned short* W2t = (unsigned short*)alloc((size_t)HID * HID * 2);
  unsigned short* Wft = (unsigned short*)alloc((size_t)OUT_P * HID * 2);

  // agg must start zeroed (combine re-zeroes it between steps)
  hipMemsetAsync(agg, 0, (size_t)N * HID * 4, stream);

  prep_weights_kernel<<<(HID * IN_F + 255) / 256, 256, 0, stream>>>(
      W1, W2, Wf, W1t, W2t, Wft);

  deg_init_kernel<<<(N + 255) / 256, 256, 0, stream>>>(deg, N);
  deg_acc_kernel<<<(E + 255) / 256, 256, 0, stream>>>(colIdx, deg, E);
  dinv_kernel<<<(N + 255) / 256, 256, 0, stream>>>(deg, N);

  encode_kernel<<<N / 16, 128, 0, stream>>>(x, W1t, b1, W2t, b2, h0, N);

  const float* cur = h0;
  for (int k = 0; k < KSTEPS; ++k) {
    float* nxt = (k & 1) ? hB : hA;
    edge_scatter_kernel<<<(E + 3) / 4, 256, 0, stream>>>(
        rowIdx, colIdx, deg, cur, agg, E);
    combine_kernel<<<(N * HID + 255) / 256, 256, 0, stream>>>(
        cur, h0, deg, agg, nxt, N);
    cur = nxt;
  }

  decode_kernel<<<N / 16, 96, 0, stream>>>(cur, Wft, bfv, out, N);
}
